// InteractionBlockGNNLayer_85744727097465
// MI455X (gfx1250) — compile-verified, matrix operand order checked by reading, not run on hardware
//
#include <hip/hip_runtime.h>
#include <math.h>

typedef __attribute__((ext_vector_type(16))) _Float16 v16h;
typedef __attribute__((ext_vector_type(8)))  _Float16 v8h;
typedef __attribute__((ext_vector_type(4)))  _Float16 v4h;
typedef __attribute__((ext_vector_type(8)))  float    v8f;

#define LOG2F 0.69314718055994530942f
#define PI_F  3.14159265358979323846f

// K-index owned by vector-half i (ISA 7.12.2, 16-bit A/B fragment layout).
__device__ __forceinline__ int kmap16(int i, int hi) {
  int v = i >> 1, h = i & 1;
  int base = (v < 4) ? (2 * v + h) : (16 + 2 * (v - 4) + h);
  return base + (hi ? 8 : 0);
}

__device__ __forceinline__ v8f wmma16(v16h a, v16h b, v8f c) {
  return __builtin_amdgcn_wmma_f32_16x16x32_f16(false, a, false, b, (short)0, c,
                                                false, false);
}

// A lane's fragment = two contiguous 8-half runs -> two ds_load_b128.
// A: row = M (lane&15), from row-major [M][K] tile.
// B: row = N column (t*16 + lane&15), from TRANSPOSED weights [N][K].
__device__ __forceinline__ v16h ldsFrag(const _Float16* t, int row, int stride,
                                        int kbase, int hi) {
  const v8h* p = (const v8h*)(t + row * stride + kbase + hi * 8);
  v8h lo = p[0];
  v8h up = p[2];   // +16 halves
  return __builtin_shufflevector(lo, up, 0, 1, 2, 3, 4, 5, 6, 7,
                                 8, 9, 10, 11, 12, 13, 14, 15);
}

template <int KS, int NT>
__device__ __forceinline__ void gemm_frag(const _Float16* atile, int astride,
                                          const _Float16* wT, int wstride,
                                          int lane, v8f* acc) {
  int m = lane & 15, hi = lane >> 4;
#pragma unroll
  for (int ks = 0; ks < KS; ++ks) {
    v16h a = ldsFrag(atile, m, astride, ks * 32, hi);
#pragma unroll
    for (int t = 0; t < NT; ++t) {
      v16h b = ldsFrag(wT, t * 16 + m, wstride, ks * 32, hi);
      acc[t] = wmma16(a, b, acc[t]);
    }
  }
}

__device__ __forceinline__ float sspf(float x) {  // shifted softplus
  float sp = (x > 15.0f) ? x : log1pf(__expf(x));
  return sp - LOG2F;
}

__device__ __forceinline__ void st4h(_Float16* p, float4 v) {
  v4h h = {(_Float16)v.x, (_Float16)v.y, (_Float16)v.z, (_Float16)v.w};
  *(v4h*)p = h;
}

// ---------------- Kernel 1: h = x @ conv_lin1_w   [N,128]@[128,64] ----------
__global__ __launch_bounds__(256) void k_lin1(const float* __restrict__ x,
                                              const float* __restrict__ w,
                                              float* __restrict__ h, int N) {
  __shared__ __align__(16) _Float16 wgtT[64 * 128];     // transposed [N=64][K=128]
  __shared__ __align__(16) _Float16 xst[8][16 * 128];
  int tid = threadIdx.x, lane = tid & 31, wv = tid >> 5;
  for (int j = tid; j < 128 * 64; j += 256) {           // coalesced read, LDS transpose
    int k = j >> 6, n = j & 63;
    wgtT[n * 128 + k] = (_Float16)w[j];
  }
  int rowbase = blockIdx.x * 128 + wv * 16;
  const float4* x4 = (const float4*)x;
  for (int j = lane; j < 16 * 32; j += 32) {
    int r = j >> 5, c4 = j & 31;
    int grow = rowbase + r;
    float4 v = make_float4(0.f, 0.f, 0.f, 0.f);
    if (grow < N) v = x4[(size_t)grow * 32 + c4];
    st4h(&xst[wv][r * 128 + c4 * 4], v);
  }
  __syncthreads();
  v8f z = {};
  v8f acc[4];
#pragma unroll
  for (int t = 0; t < 4; ++t) acc[t] = z;
  gemm_frag<4, 4>(xst[wv], 128, wgtT, 128, lane, acc);
  int ln = lane & 15, hi = lane >> 4;
#pragma unroll
  for (int t = 0; t < 4; ++t)
#pragma unroll
    for (int r = 0; r < 8; ++r) {
      int grow = rowbase + r + 8 * hi;
      if (grow < N) h[(size_t)grow * 64 + t * 16 + ln] = acc[t][r];
    }
}

// ---------------- Kernel 2: per-edge filter MLP + gather/scatter ------------
__global__ __launch_bounds__(256) void k_edge(
    const float* __restrict__ pos, const long long* __restrict__ ei,
    const float* __restrict__ w1, const float* __restrict__ b1,
    const float* __restrict__ w2, const float* __restrict__ b2,
    const float* __restrict__ h, float* __restrict__ agg, long long E) {
  __shared__ __align__(16) _Float16 w1T[64 * 64];   // transposed [N=64][K=64], K pad 50->64
  __shared__ __align__(16) _Float16 w2T[64 * 64];   // transposed [N=64][K=64]
  __shared__ float b1s[64], b2s[64];
  __shared__ __align__(16) _Float16 hid[8][16 * 64];
  __shared__ float cw[8][16];
  __shared__ float dw[8][16];
  __shared__ int ssrc[8][16], sdst[8][16];
  int tid = threadIdx.x, lane = tid & 31, wv = tid >> 5;
  for (int j = tid; j < 64 * 64; j += 256) {
    int k = j >> 6, n = j & 63;
    w1T[n * 64 + k] = (_Float16)((k < 50) ? w1[j] : 0.0f);  // w1[j] = w1[k*64+n]
    w2T[n * 64 + k] = (_Float16)w2[j];
  }
  if (tid < 64) { b1s[tid] = b1[tid]; b2s[tid] = b2[tid]; }

  long long tilebase = ((long long)blockIdx.x * 8 + wv) * 16;
  if (lane < 16) {
    long long e = tilebase + lane;
    float dist = 1e9f; int s = 0, d = 0;
    if (e < E) {
      s = (int)ei[e]; d = (int)ei[E + e];
      float dx = pos[d * 3 + 0] - pos[s * 3 + 0];
      float dy = pos[d * 3 + 1] - pos[s * 3 + 1];
      float dz = pos[d * 3 + 2] - pos[s * 3 + 2];
      dist = sqrtf(dx * dx + dy * dy + dz * dz);
      __builtin_prefetch(&h[(size_t)s * 64], 0, 1);
    }
    dw[wv][lane] = dist;
    cw[wv][lane] = 0.5f * (__cosf(dist * (PI_F / 10.0f)) + 1.0f);
    ssrc[wv][lane] = s;
    sdst[wv][lane] = d;
  }
  __syncthreads();

  const float step = 10.0f / 49.0f;
  const float coeff = -0.5f / (step * step);
  int m = lane & 15, hi = lane >> 4, ln = m;
  float dm = dw[wv][m];
  v8f z = {};
  v8f acc[4];
#pragma unroll
  for (int t = 0; t < 4; ++t) acc[t] = z;
  // GEMM1: gaussian features synthesized directly in the WMMA A layout
#pragma unroll
  for (int ks = 0; ks < 2; ++ks) {
    v16h a;
#pragma unroll
    for (int i = 0; i < 16; ++i) {
      int k = ks * 32 + kmap16(i, hi);
      float df = dm - step * (float)k;
      float g = (k < 50) ? __expf(coeff * df * df) : 0.0f;
      a[i] = (_Float16)g;
    }
#pragma unroll
    for (int t = 0; t < 4; ++t) {
      v16h b = ldsFrag(w1T, t * 16 + m, 64, ks * 32, hi);
      acc[t] = wmma16(a, b, acc[t]);
    }
  }
#pragma unroll
  for (int t = 0; t < 4; ++t)
#pragma unroll
    for (int r = 0; r < 8; ++r) {
      int mm = r + 8 * hi, n = t * 16 + ln;
      hid[wv][mm * 64 + n] = (_Float16)sspf(acc[t][r] + b1s[n]);
    }
  __syncthreads();
  // GEMM2: hidden @ mlp_w2
#pragma unroll
  for (int t = 0; t < 4; ++t) acc[t] = z;
  gemm_frag<2, 4>(hid[wv], 64, w2T, 64, lane, acc);
  // W *= C; msg = h[src]*W; atomic scatter-add to agg[dst]
#pragma unroll
  for (int t = 0; t < 4; ++t)
#pragma unroll
    for (int r = 0; r < 8; ++r) {
      int mm = r + 8 * hi, n = t * 16 + ln;
      if (tilebase + mm < E) {
        float wval = (acc[t][r] + b2s[n]) * cw[wv][mm];
        float hv = h[(size_t)ssrc[wv][mm] * 64 + n];
        atomicAdd(&agg[(size_t)sdst[wv][mm] * 64 + n], hv * wval);
      }
    }
}

// ---------------- Kernel 3: fused node tail --------------------------------
__global__ __launch_bounds__(128) void k_tail(
    const float* __restrict__ x, const float* __restrict__ agg,
    const float* __restrict__ w2, const float* __restrict__ b2,
    const float* __restrict__ wi, const float* __restrict__ bi,
    const float* __restrict__ wl, const float* __restrict__ bl,
    float* __restrict__ out, int N) {
  __shared__ __align__(16) _Float16 wst[128 * 128];  // staging: w2T / wiT / wlT
  __shared__ __align__(16) _Float16 sc[4][16 * 128];
  __shared__ float b2s[128], bis[128], bls[128];
  int tid = threadIdx.x, lane = tid & 31, wv = tid >> 5;
  if (tid < 128) { b2s[tid] = b2[tid]; bis[tid] = bi[tid]; bls[tid] = bl[tid]; }
  // stage conv_lin2_w [64,128] transposed -> [N=128][K=64], stride 64
  for (int j = tid; j < 64 * 128; j += 128) {
    int k = j >> 7, n = j & 127;
    wst[n * 64 + k] = (_Float16)w2[j];
  }
  int rowbase = blockIdx.x * 64 + wv * 16;
  const float4* a4 = (const float4*)agg;
  for (int j = lane; j < 16 * 16; j += 32) {
    int r = j >> 4, c4 = j & 15;
    int grow = rowbase + r;
    float4 v = make_float4(0.f, 0.f, 0.f, 0.f);
    if (grow < N) v = a4[(size_t)grow * 16 + c4];
    st4h(&sc[wv][r * 128 + c4 * 4], v);
  }
  __syncthreads();
  int ln = lane & 15, hi = lane >> 4;
  v8f z = {};
  v8f acc[8];
  // ---- h2 = ssp(agg @ conv_lin2_w + b)  (K=64)
#pragma unroll
  for (int t = 0; t < 8; ++t) acc[t] = z;
  gemm_frag<2, 8>(sc[wv], 128, wst, 64, lane, acc);
#pragma unroll
  for (int t = 0; t < 8; ++t)
#pragma unroll
    for (int r = 0; r < 8; ++r) {
      int mm = r + 8 * hi, n = t * 16 + ln;
      sc[wv][mm * 128 + n] = (_Float16)sspf(acc[t][r] + b2s[n]);
    }
  __syncthreads();
  // stage int_lin_w [128,128] transposed, stride 128
  for (int j = tid; j < 128 * 128; j += 128) {
    int k = j >> 7, n = j & 127;
    wst[n * 128 + k] = (_Float16)wi[j];
  }
  __syncthreads();
  // ---- h3 = h2 @ int_lin_w + b  (K=128)
#pragma unroll
  for (int t = 0; t < 8; ++t) acc[t] = z;
  gemm_frag<4, 8>(sc[wv], 128, wst, 128, lane, acc);
#pragma unroll
  for (int t = 0; t < 8; ++t)
#pragma unroll
    for (int r = 0; r < 8; ++r) {
      int mm = r + 8 * hi, n = t * 16 + ln;
      sc[wv][mm * 128 + n] = (_Float16)(acc[t][r] + bis[n]);
    }
  __syncthreads();
  // stage lin1_w [128,128] transposed
  for (int j = tid; j < 128 * 128; j += 128) {
    int k = j >> 7, n = j & 127;
    wst[n * 128 + k] = (_Float16)wl[j];
  }
  __syncthreads();
  // ---- y = relu(h3 @ lin1_w + b); out = x + y  (K=128)
#pragma unroll
  for (int t = 0; t < 8; ++t) acc[t] = z;
  gemm_frag<4, 8>(sc[wv], 128, wst, 128, lane, acc);
#pragma unroll
  for (int t = 0; t < 8; ++t)
#pragma unroll
    for (int r = 0; r < 8; ++r) {
      int mm = r + 8 * hi, n = t * 16 + ln;
      int grow = rowbase + mm;
      if (grow < N) {
        float y = fmaxf(acc[t][r] + bls[n], 0.0f);
        out[(size_t)grow * 128 + n] = x[(size_t)grow * 128 + n] + y;
      }
    }
}

extern "C" void kernel_launch(void* const* d_in, const int* in_sizes, int n_in,
                              void* d_out, int out_size, void* d_ws, size_t ws_size,
                              hipStream_t stream) {
  const float*     x      = (const float*)d_in[0];
  const float*     pos    = (const float*)d_in[1];
  const long long* ei     = (const long long*)d_in[2];
  const float*     mlp_w1 = (const float*)d_in[3];
  const float*     mlp_b1 = (const float*)d_in[4];
  const float*     mlp_w2 = (const float*)d_in[5];
  const float*     mlp_b2 = (const float*)d_in[6];
  const float*     cl1w   = (const float*)d_in[7];
  const float*     cl2w   = (const float*)d_in[8];
  const float*     cl2b   = (const float*)d_in[9];
  const float*     intw   = (const float*)d_in[10];
  const float*     intb   = (const float*)d_in[11];
  const float*     l1w    = (const float*)d_in[12];
  const float*     l1b    = (const float*)d_in[13];

  int       N = in_sizes[0] / 128;
  long long E = (long long)in_sizes[2] / 2;

  float* hbuf = (float*)d_ws;                 // [N,64]
  float* agg  = hbuf + (size_t)N * 64;        // [N,64]
  hipMemsetAsync(agg, 0, (size_t)N * 64 * sizeof(float), stream);

  k_lin1<<<(N + 127) / 128, 256, 0, stream>>>(x, cl1w, hbuf, N);
  k_edge<<<(int)((E + 127) / 128), 256, 0, stream>>>(pos, ei, mlp_w1, mlp_b1,
                                                     mlp_w2, mlp_b2, hbuf, agg, E);
  k_tail<<<(N + 63) / 64, 128, 0, stream>>>(x, agg, cl2w, cl2b, intw, intb,
                                            l1w, l1b, (float*)d_out, N);
}
